// Block_85864986181683
// MI455X (gfx1250) — matrix operand order repaired and checked by reading, not compiled, not run on hardware
//
#include <hip/hip_runtime.h>
#include <math.h>

// ---------------- problem constants ----------------
#define N_EMB 512
#define HEADS 8
#define HS    64
#define BB    4
#define TSEQ  2048
#define ROWS  (BB * TSEQ)   // 8192
#define EPSLN 1e-5f

typedef __attribute__((ext_vector_type(16))) __bf16 v16bf;
typedef __attribute__((ext_vector_type(8)))  float  v8f;

union AFragQ { v16bf v; uint4 q[2]; };

// ---------------- CDNA5 async global->LDS (guarded) ----------------
#if defined(__AMDGCN__) && __has_builtin(__builtin_amdgcn_global_load_async_to_lds_b128)
#define ASYNC_LDS 1
#else
#define ASYNC_LDS 0
#endif

typedef int v4i_gcc __attribute__((vector_size(16)));
typedef __attribute__((address_space(1))) v4i_gcc* as1_v4i;
typedef __attribute__((address_space(3))) v4i_gcc* as3_v4i;

__device__ __forceinline__ void async_b128(const __bf16* g, __bf16* l) {
#if ASYNC_LDS
  __builtin_amdgcn_global_load_async_to_lds_b128((as1_v4i)g, (as3_v4i)l, 0, 0);
#else
  *(uint4*)l = *(const uint4*)g;   // fallback: register round-trip
#endif
}

template <int N>
__device__ __forceinline__ void wait_async() {
#if defined(__AMDGCN__) && __has_builtin(__builtin_amdgcn_s_wait_asynccnt)
  __builtin_amdgcn_s_wait_asynccnt((unsigned short)N);
#elif ASYNC_LDS
  asm volatile("s_wait_asynccnt %0" ::"i"(N) : "memory");
#endif
}

__device__ __forceinline__ v8f wmma_bf16(v16bf a, v16bf b, v8f c) {
  // D = A(16x32 bf16) x B(32x16 bf16) + C(16x16 f32)
  return __builtin_amdgcn_wmma_f32_16x16x32_bf16(
      /*neg_a=*/false, a, /*neg_b=*/false, b,
      /*c_mod=*/(short)0, c, /*reuse_a=*/false, /*reuse_b=*/false);
}

// A fragment: 16x32 bf16 tile, row-major [M][K], ld in elements (mult of 16).
// ISA layout: lane l (0-15) = row M=l; VGPR 0-3 hold K = 8g+0..7 (g = lane>>4),
// VGPR 4-7 hold K = 16+8g+0..7  ->  two contiguous 16B loads per lane.
__device__ __forceinline__ v16bf load_a_frag(const __bf16* base, int ld, int lane) {
  const int l = lane & 15, g = lane >> 4;
  const __bf16* rp = base + (size_t)l * ld;
  AFragQ f;
  f.q[0] = *(const uint4*)(rp + 8 * g);
  f.q[1] = *(const uint4*)(rp + 16 + 8 * g);
  return f.v;
}

// B fragment: 32x16 bf16 tile, from an N-major source [N][K], ld in elements.
// Layout: lane l = column N=l; VGPR 0-3 hold K = 8g+0..7 (g = lane>>4),
// VGPR 4-7 hold K = 16+8g+0..7  ->  two contiguous 16B loads per lane.
__device__ __forceinline__ v16bf load_b_frag(const __bf16* base, int ld, int lane) {
  const int l = lane & 15, g = lane >> 4;
  const __bf16* rp = base + (size_t)l * ld;
  AFragQ f;
  f.q[0] = *(const uint4*)(rp + 8 * g);
  f.q[1] = *(const uint4*)(rp + 16 + 8 * g);
  return f.v;
}

// ---------------- LayerNorm 1: x -> bf16 h ----------------
__global__ void __launch_bounds__(256)
k_ln1(const float* __restrict__ x, const float* __restrict__ gw,
      const float* __restrict__ bw, __bf16* __restrict__ h) {
  const int lane = threadIdx.x & 31;
  const int row  = blockIdx.x * 8 + (threadIdx.x >> 5);
  const float* xr = x + (size_t)row * N_EMB + lane * 16;
  float v[16];
  float s = 0.f, ss = 0.f;
#pragma unroll
  for (int i = 0; i < 16; ++i) { v[i] = xr[i]; s += v[i]; ss += v[i] * v[i]; }
#pragma unroll
  for (int m = 16; m >= 1; m >>= 1) {
    s  += __shfl_xor(s,  m, 32);
    ss += __shfl_xor(ss, m, 32);
  }
  const float mu   = s * (1.f / N_EMB);
  const float var  = ss * (1.f / N_EMB) - mu * mu;
  const float rstd = rsqrtf(var + EPSLN);
  const float* gp = gw + lane * 16;
  const float* bp = bw + lane * 16;
  __bf16* hr = h + (size_t)row * N_EMB + lane * 16;
#pragma unroll
  for (int i = 0; i < 16; ++i)
    hr[i] = (__bf16)((v[i] - mu) * rstd * gp[i] + bp[i]);
}

// ---------------- residual add + LayerNorm 2 ----------------
__global__ void __launch_bounds__(256)
k_add_ln2(const float* __restrict__ x, float* __restrict__ x1,
          const float* __restrict__ gw, const float* __restrict__ bw,
          __bf16* __restrict__ h2) {
  const int lane = threadIdx.x & 31;
  const int row  = blockIdx.x * 8 + (threadIdx.x >> 5);
  const float* xr = x + (size_t)row * N_EMB + lane * 16;
  float* x1r = x1 + (size_t)row * N_EMB + lane * 16;
  float v[16];
  float s = 0.f, ss = 0.f;
#pragma unroll
  for (int i = 0; i < 16; ++i) {
    v[i] = xr[i] + x1r[i];
    x1r[i] = v[i];
    s += v[i]; ss += v[i] * v[i];
  }
#pragma unroll
  for (int m = 16; m >= 1; m >>= 1) {
    s  += __shfl_xor(s,  m, 32);
    ss += __shfl_xor(ss, m, 32);
  }
  const float mu   = s * (1.f / N_EMB);
  const float var  = ss * (1.f / N_EMB) - mu * mu;
  const float rstd = rsqrtf(var + EPSLN);
  const float* gp = gw + lane * 16;
  const float* bp = bw + lane * 16;
  __bf16* hr = h2 + (size_t)row * N_EMB + lane * 16;
#pragma unroll
  for (int i = 0; i < 16; ++i)
    hr[i] = (__bf16)((v[i] - mu) * rstd * gp[i] + bp[i]);
}

// ---------------- weight conversions (fp32 -> bf16, transposed to N-major) ----
// in: [H][C][d] fp32   out: [H][d][C] bf16
__global__ void __launch_bounds__(256)
k_cvt_wqkv_t(const float* __restrict__ in, __bf16* __restrict__ out) {
  const int i = blockIdx.x * blockDim.x + threadIdx.x;   // i = (h*HS + d)*N_EMB + c
  const int c  = i & (N_EMB - 1);
  const int hd = i >> 9;
  const int d  = hd & (HS - 1);
  const int hh = hd >> 6;
  out[i] = (__bf16)in[((size_t)hh * N_EMB + c) * HS + d];
}

// in: [C][N] fp32   out: [N][C] bf16
__global__ void __launch_bounds__(256)
k_cvt_wff_t(const float* __restrict__ in, __bf16* __restrict__ out) {
  const int i = blockIdx.x * blockDim.x + threadIdx.x;   // i = n*N_EMB + c
  const int c = i & (N_EMB - 1);
  const int n = i >> 9;
  out[i] = (__bf16)in[(size_t)c * N_EMB + n];
}

// ---------------- QKV projection GEMM ----------------
// block = one 16-row M-tile of one matrix (q/k/v); wave w = head w.
// A slab (16x32 per K-step) staged once per block via async copy, double-buffered.
__global__ void __launch_bounds__(256)
k_qkv(const __bf16* __restrict__ h,
      const __bf16* __restrict__ wqt, const __bf16* __restrict__ wkt,
      const __bf16* __restrict__ wvt,
      __bf16* __restrict__ Q, __bf16* __restrict__ K, __bf16* __restrict__ Vt) {
  __shared__ __align__(16) __bf16 ab[2][16 * 32];
  const int tid  = threadIdx.x;
  const int lane = tid & 31;
  const int hh   = tid >> 5;                 // wave = head
  const int mt   = blockIdx.x / 3;
  const int mat  = blockIdx.x % 3;
  const int row0 = mt * 16;
  const __bf16* Wt = (mat == 0 ? wqt : mat == 1 ? wkt : wvt) + (size_t)hh * HS * N_EMB;
  const __bf16* A  = h + (size_t)row0 * N_EMB;
  const int cr = tid >> 2, cc = (tid & 3) * 8;   // copy lanes: 64 x b128

  if (tid < 64) async_b128(A + (size_t)cr * N_EMB + cc, &ab[0][cr * 32 + cc]);

  v8f a0 = {}, a1 = {}, a2 = {}, a3 = {};
  for (int i = 0; i < 16; ++i) {
    const int kk = i * 32;
    if (i + 1 < 16) {
      if (tid < 64)
        async_b128(A + (size_t)cr * N_EMB + kk + 32 + cc, &ab[(i + 1) & 1][cr * 32 + cc]);
      wait_async<1>();
    } else {
      wait_async<0>();
    }
    __syncthreads();
    const v16bf a  = load_a_frag(&ab[i & 1][0], 32, lane);   // ds_load_b128 x2
    __builtin_prefetch(Wt + kk + 64, 0, 3);
    const v16bf b0 = load_b_frag(Wt + kk,                      N_EMB, lane);
    const v16bf b1 = load_b_frag(Wt + (size_t)16 * N_EMB + kk, N_EMB, lane);
    const v16bf b2 = load_b_frag(Wt + (size_t)32 * N_EMB + kk, N_EMB, lane);
    const v16bf b3 = load_b_frag(Wt + (size_t)48 * N_EMB + kk, N_EMB, lane);
    a0 = wmma_bf16(a, b0, a0);
    a1 = wmma_bf16(a, b1, a1);
    a2 = wmma_bf16(a, b2, a2);
    a3 = wmma_bf16(a, b3, a3);
    __syncthreads();
  }
  const int l = lane & 15, g = lane >> 4;
  const float scl = (mat == 0) ? 0.125f : 1.0f;   // fold d^-1/2 into Q
#pragma unroll
  for (int r = 0; r < 8; ++r) {
    const int m  = row0 + r + 8 * g;
    const int b_ = m >> 11, t = m & (TSEQ - 1);
    const int bh = b_ * HEADS + hh;
    const float e0 = a0[r] * scl, e1 = a1[r] * scl, e2 = a2[r] * scl, e3 = a3[r] * scl;
    if (mat == 2) {
      __bf16* vp = Vt + ((size_t)bh * HS) * TSEQ + t;      // Vt[bh][d][t]
      vp[(size_t)(l) * TSEQ]      = (__bf16)e0;
      vp[(size_t)(16 + l) * TSEQ] = (__bf16)e1;
      vp[(size_t)(32 + l) * TSEQ] = (__bf16)e2;
      vp[(size_t)(48 + l) * TSEQ] = (__bf16)e3;
    } else {
      __bf16* op = (mat == 0 ? Q : K) + ((size_t)bh * TSEQ + t) * HS;
      op[l]      = (__bf16)e0;
      op[16 + l] = (__bf16)e1;
      op[32 + l] = (__bf16)e2;
      op[48 + l] = (__bf16)e3;
    }
  }
}

// ---------------- flash attention ----------------
// one wave per 16-query tile per (b,h); online softmax; P staged via LDS.
// K is [t][d] (N-major for QK^T); V is transposed [d][t] (N-major for P*V).
__global__ void __launch_bounds__(256)
k_attn(const __bf16* __restrict__ Q, const __bf16* __restrict__ K,
       const __bf16* __restrict__ Vt, float* __restrict__ sa) {
  __shared__ __align__(16) __bf16 pbuf[8][16 * 32];
  const int lane = threadIdx.x & 31;
  const int wib  = threadIdx.x >> 5;
  const int wid  = blockIdx.x * 8 + wib;
  const int bh   = wid >> 7;            // b*HEADS + h
  const int t0   = (wid & 127) * 16;
  const int b_   = bh >> 3, hh = bh & 7;
  const __bf16* Qp = Q + ((size_t)bh * TSEQ + t0) * HS;
  const __bf16* Kp = K + (size_t)bh * TSEQ * HS;
  const __bf16* Vp = Vt + (size_t)bh * HS * TSEQ;
  const int l = lane & 15, g = lane >> 4;

  const v16bf qa0 = load_a_frag(Qp, HS, lane);        // d = 0..31
  const v16bf qa1 = load_a_frag(Qp + 32, HS, lane);   // d = 32..63

  v8f o0 = {}, o1 = {}, o2 = {}, o3 = {};
  float rmax[8], rsum[8];
#pragma unroll
  for (int r = 0; r < 8; ++r) { rmax[r] = -__builtin_inff(); rsum[r] = 0.f; }

  const int send = t0 + 15;
  __bf16* pb = pbuf[wib];

  for (int s0 = 0; s0 <= send; s0 += 32) {
    // --- scores for key tiles [s0, s0+16) and [s0+16, s0+32) ---
    v8f sf0 = {}, sf1 = {};
    {
      const v16bf b0 = load_b_frag(Kp + (size_t)s0 * HS,      HS, lane);
      const v16bf b1 = load_b_frag(Kp + (size_t)s0 * HS + 32, HS, lane);
      sf0 = wmma_bf16(qa0, b0, sf0);
      sf0 = wmma_bf16(qa1, b1, sf0);
    }
    const bool v1 = (s0 + 16 <= send);  // wave-uniform
    if (v1) {
      const v16bf b0 = load_b_frag(Kp + (size_t)(s0 + 16) * HS,      HS, lane);
      const v16bf b1 = load_b_frag(Kp + (size_t)(s0 + 16) * HS + 32, HS, lane);
      sf1 = wmma_bf16(qa0, b0, sf1);
      sf1 = wmma_bf16(qa1, b1, sf1);
    }
    // --- causal mask + online softmax (row = r + 8g, key col = lane&15) ---
#pragma unroll
    for (int r = 0; r < 8; ++r) {
      const int tg = t0 + r + 8 * g;
      float e0 = sf0[r], e1 = sf1[r];
      if (s0 + l > tg) e0 = -__builtin_inff();
      if (!v1 || (s0 + 16 + l > tg)) e1 = -__builtin_inff();
      float mx = fmaxf(e0, e1);
      mx = fmaxf(mx, __shfl_xor(mx, 1, 32));
      mx = fmaxf(mx, __shfl_xor(mx, 2, 32));
      mx = fmaxf(mx, __shfl_xor(mx, 4, 32));
      mx = fmaxf(mx, __shfl_xor(mx, 8, 32));
      const float nmax = fmaxf(rmax[r], mx);
      const float corr = __expf(rmax[r] - nmax);   // 0 on first chunk
      const float p0 = __expf(e0 - nmax);
      const float p1 = __expf(e1 - nmax);
      float ps = p0 + p1;
      ps += __shfl_xor(ps, 1, 32);
      ps += __shfl_xor(ps, 2, 32);
      ps += __shfl_xor(ps, 4, 32);
      ps += __shfl_xor(ps, 8, 32);
      rsum[r] = rsum[r] * corr + ps;
      rmax[r] = nmax;
      o0[r] = o0[r] * corr; o1[r] = o1[r] * corr;
      o2[r] = o2[r] * corr; o3[r] = o3[r] * corr;
      const int m = r + 8 * g;
      pb[m * 32 + l]      = (__bf16)p0;   // P tile row-major 16x32 in LDS
      pb[m * 32 + 16 + l] = (__bf16)p1;
    }
    // --- P (from LDS, A-layout, ds_load_b128 x2) x V^T (N-major) ---
    const v16bf pa  = load_a_frag(pb, 32, lane);
    const v16bf vb0 = load_b_frag(Vp + s0,                      TSEQ, lane);
    const v16bf vb1 = load_b_frag(Vp + (size_t)16 * TSEQ + s0,  TSEQ, lane);
    const v16bf vb2 = load_b_frag(Vp + (size_t)32 * TSEQ + s0,  TSEQ, lane);
    const v16bf vb3 = load_b_frag(Vp + (size_t)48 * TSEQ + s0,  TSEQ, lane);
    o0 = wmma_bf16(pa, vb0, o0);
    o1 = wmma_bf16(pa, vb1, o1);
    o2 = wmma_bf16(pa, vb2, o2);
    o3 = wmma_bf16(pa, vb3, o3);
  }
  // --- normalize and write concatenated-head output sa[b][t][h*64 + d] ---
  float* outp = sa + ((size_t)b_ * TSEQ + t0) * N_EMB + hh * HS;
#pragma unroll
  for (int r = 0; r < 8; ++r) {
    const int m = r + 8 * g;
    const float inv = 1.0f / rsum[r];
    float* orow = outp + (size_t)m * N_EMB;
    orow[l]      = o0[r] * inv;
    orow[16 + l] = o1[r] * inv;
    orow[32 + l] = o2[r] * inv;
    orow[48 + l] = o3[r] * inv;
  }
}

// ---------------- feed-forward GEMM + bias + ReLU + residual ----------------
// block = one 16-row M-tile; wave w = N-columns [64w, 64w+64).
// A slab staged once per block via async copy, double-buffered.
__global__ void __launch_bounds__(256)
k_ff(const __bf16* __restrict__ h2, const __bf16* __restrict__ wft,
     const float* __restrict__ bias, const float* __restrict__ x1,
     float* __restrict__ out) {
  __shared__ __align__(16) __bf16 ab[2][16 * 32];
  const int tid  = threadIdx.x;
  const int lane = tid & 31;
  const int n0   = (tid >> 5) * 64;          // wave -> N-group
  const int row0 = blockIdx.x * 16;
  const __bf16* A = h2 + (size_t)row0 * N_EMB;
  const __bf16* W = wft + (size_t)n0 * N_EMB;
  const int cr = tid >> 2, cc = (tid & 3) * 8;

  if (tid < 64) async_b128(A + (size_t)cr * N_EMB + cc, &ab[0][cr * 32 + cc]);

  v8f a0 = {}, a1 = {}, a2 = {}, a3 = {};
  for (int i = 0; i < 16; ++i) {
    const int kk = i * 32;
    if (i + 1 < 16) {
      if (tid < 64)
        async_b128(A + (size_t)cr * N_EMB + kk + 32 + cc, &ab[(i + 1) & 1][cr * 32 + cc]);
      wait_async<1>();
    } else {
      wait_async<0>();
    }
    __syncthreads();
    const v16bf a  = load_a_frag(&ab[i & 1][0], 32, lane);
    __builtin_prefetch(W + kk + 64, 0, 3);
    const v16bf b0 = load_b_frag(W + kk,                      N_EMB, lane);
    const v16bf b1 = load_b_frag(W + (size_t)16 * N_EMB + kk, N_EMB, lane);
    const v16bf b2 = load_b_frag(W + (size_t)32 * N_EMB + kk, N_EMB, lane);
    const v16bf b3 = load_b_frag(W + (size_t)48 * N_EMB + kk, N_EMB, lane);
    a0 = wmma_bf16(a, b0, a0);
    a1 = wmma_bf16(a, b1, a1);
    a2 = wmma_bf16(a, b2, a2);
    a3 = wmma_bf16(a, b3, a3);
    __syncthreads();
  }
  const int l = lane & 15, g = lane >> 4;
  const float bs0 = bias[n0 + l],      bs1 = bias[n0 + 16 + l];
  const float bs2 = bias[n0 + 32 + l], bs3 = bias[n0 + 48 + l];
#pragma unroll
  for (int r = 0; r < 8; ++r) {
    const int m = row0 + r + 8 * g;
    const size_t idx = (size_t)m * N_EMB + n0 + l;
    float v0 = a0[r] + bs0, v1 = a1[r] + bs1, v2 = a2[r] + bs2, v3 = a3[r] + bs3;
    v0 = v0 > 0.f ? v0 : 0.f;
    v1 = v1 > 0.f ? v1 : 0.f;
    v2 = v2 > 0.f ? v2 : 0.f;
    v3 = v3 > 0.f ? v3 : 0.f;
    out[idx]      = x1[idx]      + v0;
    out[idx + 16] = x1[idx + 16] + v1;
    out[idx + 32] = x1[idx + 32] + v2;
    out[idx + 48] = x1[idx + 48] + v3;
  }
}

// ---------------- launcher ----------------
extern "C" void kernel_launch(void* const* d_in, const int* in_sizes, int n_in,
                              void* d_out, int out_size, void* d_ws, size_t ws_size,
                              hipStream_t stream) {
  (void)in_sizes; (void)n_in; (void)out_size; (void)ws_size;
  const float* x   = (const float*)d_in[0];
  const float* wq  = (const float*)d_in[1];
  const float* wk  = (const float*)d_in[2];
  const float* wv  = (const float*)d_in[3];
  const float* wff = (const float*)d_in[4];
  const float* bff = (const float*)d_in[5];
  const float* g1  = (const float*)d_in[6];
  const float* b1  = (const float*)d_in[7];
  const float* g2  = (const float*)d_in[8];
  const float* b2  = (const float*)d_in[9];

  // workspace layout (bf16 unless noted); total ~50 MB
  const size_t ACT = (size_t)ROWS * N_EMB;          // 4,194,304 elements
  const size_t WSZ = (size_t)HEADS * N_EMB * HS;    //   262,144 elements
  __bf16* hb  = (__bf16*)d_ws;     // h (LN1 out) / later h2 (LN2 out)
  __bf16* wqt = hb  + ACT;         // transposed bf16 weights [h][d][c]
  __bf16* wkt = wqt + WSZ;
  __bf16* wvt = wkt + WSZ;
  __bf16* wft = wvt + WSZ;         // [n][c]
  __bf16* Qb  = wft + (size_t)N_EMB * N_EMB;
  __bf16* Kb  = Qb + ACT;
  __bf16* Vtb = Kb + ACT;          // V transposed [bh][d][t]
  float*  sa  = (float*)(Vtb + ACT);  // attention out, then x1 = x + sa

  k_ln1<<<ROWS / 8, 256, 0, stream>>>(x, g1, b1, hb);

  k_cvt_wqkv_t<<<(int)(WSZ / 256), 256, 0, stream>>>(wq, wqt);
  k_cvt_wqkv_t<<<(int)(WSZ / 256), 256, 0, stream>>>(wk, wkt);
  k_cvt_wqkv_t<<<(int)(WSZ / 256), 256, 0, stream>>>(wv, wvt);
  k_cvt_wff_t<<<N_EMB * N_EMB / 256, 256, 0, stream>>>(wff, wft);

  // 512 M-tiles x 3 matrices; 8 waves = 8 heads per block
  k_qkv<<<1536, 256, 0, stream>>>(hb, wqt, wkt, wvt, Qb, Kb, Vtb);

  // 4 * 8 * 128 = 4096 waves -> 512 blocks
  k_attn<<<512, 256, 0, stream>>>(Qb, Kb, Vtb, sa);

  k_add_ln2<<<ROWS / 8, 256, 0, stream>>>(x, sa, g2, b2, hb);

  // 512 M-tiles; 8 waves = 8 N-groups per block
  k_ff<<<512, 256, 0, stream>>>(hb, wft, bff, sa, (float*)d_out);
}
